// KVPrefixAttention_26182120636596
// MI455X (gfx1250) — matrix operand order repaired and checked by reading, not compiled, not run on hardware
//
#include <hip/hip_runtime.h>

// KV-prefix attention for MI455X (gfx1250, wave32, WMMA).
// B=4, H=12, S=2048, D=64, HID=768, G=256, SCALE=8.
// Outputs: context [B,H,S,D] f32, then weights [B,H,S,S+1] f32 (flat concat).

typedef __attribute__((ext_vector_type(16))) __bf16 v16bf;
typedef __attribute__((ext_vector_type(8)))  float  v8f;

#define WMMA_BF16(a, b, c) \
  __builtin_amdgcn_wmma_f32_16x16x32_bf16(false, (a), false, (b), (short)0, (c), false, false)

__device__ __forceinline__ float rmax16(float v) {
  v = fmaxf(v, __shfl_xor(v, 1, 32));
  v = fmaxf(v, __shfl_xor(v, 2, 32));
  v = fmaxf(v, __shfl_xor(v, 4, 32));
  v = fmaxf(v, __shfl_xor(v, 8, 32));
  return v;
}
__device__ __forceinline__ float rsum16(float v) {
  v += __shfl_xor(v, 1, 32);
  v += __shfl_xor(v, 2, 32);
  v += __shfl_xor(v, 4, 32);
  v += __shfl_xor(v, 8, 32);
  return v;
}

// 16 consecutive f32 -> bf16 fragment elements 0..15 (B-matrix lane slice).
__device__ __forceinline__ v16bf load16bf(const float* p) {
  v16bf r;
  const float4* q = (const float4*)p;
#pragma unroll
  for (int i = 0; i < 4; ++i) {
    float4 x = q[i];
    r[4 * i + 0] = (__bf16)x.x; r[4 * i + 1] = (__bf16)x.y;
    r[4 * i + 2] = (__bf16)x.z; r[4 * i + 3] = (__bf16)x.w;
  }
  return r;
}

// A-matrix lane slice: 8 floats at p -> elems 0..7, 8 floats at p+16 -> elems 8..15.
__device__ __forceinline__ v16bf loadA2run(const float* p) {
  v16bf r;
  const float4* q0 = (const float4*)p;
  const float4* q1 = (const float4*)(p + 16);
#pragma unroll
  for (int i = 0; i < 2; ++i) {
    float4 x = q0[i];
    r[4 * i + 0] = (__bf16)x.x; r[4 * i + 1] = (__bf16)x.y;
    r[4 * i + 2] = (__bf16)x.z; r[4 * i + 3] = (__bf16)x.w;
    float4 y = q1[i];
    r[8 + 4 * i + 0] = (__bf16)y.x; r[8 + 4 * i + 1] = (__bf16)y.y;
    r[8 + 4 * i + 2] = (__bf16)y.z; r[8 + 4 * i + 3] = (__bf16)y.w;
  }
  return r;
}

__global__ __launch_bounds__(128) void kvprefix_attn_kernel(
    const float* __restrict__ Q, const float* __restrict__ K,
    const float* __restrict__ V, const float* __restrict__ GS,
    const float* __restrict__ MASK, const float* __restrict__ WK,
    const float* __restrict__ BK, const float* __restrict__ WV,
    const float* __restrict__ BV,
    float* __restrict__ OUTC, float* __restrict__ OUTW) {
  __shared__ __align__(16) __bf16 Wbuf[4][16][32];  // per-wave prob tile (C->A relayout)
  __shared__ float Sdot[4][16];                     // per-wave prefix-score broadcast
  __shared__ float PKV[2][64];                      // prefix K (0) / V (1) for this (b,h)

  const int tid   = threadIdx.x;
  const int lane  = tid & 31;
  const int wave  = tid >> 5;
  const int lanel = lane & 15;
  const int laneh = lane >> 4;

  const int bid  = blockIdx.x;  // 768 blocks = 48 (b,h) * 16 row-blocks
  const int bh   = bid >> 4;
  const int rblk = bid & 15;
  const int b    = bh / 12;
  const int h    = bh % 12;

  // ---- Prefix token projection: PKV[w][d] = GS[b,:] @ W[:, h*64+d] + bias ----
  {
    int which = tid >> 6;  // threads 0..63 -> K, 64..127 -> V
    int d     = tid & 63;
    int o     = h * 64 + d;
    const float* W  = which ? WV : WK;
    float acc       = which ? BV[o] : BK[o];
    const float* g  = GS + b * 256;
#pragma unroll 4
    for (int gg = 0; gg < 256; ++gg) acc += g[gg] * W[gg * 768 + o];
    PKV[which][d] = acc;
  }
  __syncthreads();

  const size_t bhS = (size_t)bh * 2048;
  const float* Qb  = Q + bhS * 64;
  const float* Kb  = K + bhS * 64;
  const float* Vb  = V + bhS * 64;
  const float* Mb  = MASK + (size_t)b * 2048;

  const int q0 = rblk * 128 + wave * 32;  // this wave owns rows q0..q0+31

  // ---- Q as bf16 A-fragments (2 row-tiles x 2 k-halves), held in VGPRs ----
  v16bf aq[2][2];
#pragma unroll
  for (int t = 0; t < 2; ++t) {
    const float* rowp = Qb + (size_t)(q0 + t * 16 + lanel) * 64;
#pragma unroll
    for (int f = 0; f < 2; ++f)
      aq[t][f] = loadA2run(rowp + f * 32 + (laneh ? 8 : 0));
  }

  // ---- Prefix scores: s_pre[row] = (Q[row,:] . prefK) / 8  (replicated to C-layout) ----
  float sp[2][8];
#pragma unroll
  for (int t = 0; t < 2; ++t) {
    float part = 0.f;
#pragma unroll
    for (int f = 0; f < 2; ++f) {
#pragma unroll
      for (int e = 0; e < 16; ++e) {
        int d = f * 32 + (laneh ? 8 : 0) + (e < 8 ? e : e + 8);
        part += (float)aq[t][f][e] * PKV[0][d];
      }
    }
    float dot = part + __shfl_xor(part, 16, 32);  // combine complementary halves
    if (!laneh) Sdot[wave][lanel] = dot * 0.125f;
    __syncthreads();
#pragma unroll
    for (int r = 0; r < 8; ++r) sp[t][r] = Sdot[wave][r + laneh * 8];
    __syncthreads();
  }

  // ---- Online-softmax state; prefix token seeds it (m=s_pre, l=1, ctx=prefV) ----
  float m_[2][8], l_[2][8];
  v8f   cacc[2][4];
  float pvv[4];
#pragma unroll
  for (int s4 = 0; s4 < 4; ++s4) pvv[s4] = PKV[1][s4 * 16 + lanel];
#pragma unroll
  for (int t = 0; t < 2; ++t)
#pragma unroll
    for (int r = 0; r < 8; ++r) { m_[t][r] = sp[t][r]; l_[t][r] = 1.0f; }
#pragma unroll
  for (int t = 0; t < 2; ++t)
#pragma unroll
    for (int s4 = 0; s4 < 4; ++s4)
#pragma unroll
      for (int r = 0; r < 8; ++r) cacc[t][s4][r] = pvv[s4];

  // ================= PASS 1: flash attention (stats + context) =================
  for (int kb = 0; kb < 64; ++kb) {
    const int key0 = kb * 32;

    v16bf bkf[2][2];  // K^T B-fragments: [key-tile j][d-half f]
#pragma unroll
    for (int j = 0; j < 2; ++j)
#pragma unroll
      for (int f = 0; f < 2; ++f)
        bkf[j][f] = loadA2run(Kb + (size_t)(key0 + j * 16 + lanel) * 64 + f * 32 + (laneh ? 8 : 0));

    v16bf bvf[4];  // V B-fragments: [dim-slice]; elem e -> key = key0+(e<8?e:e+8)+8*laneh
#pragma unroll
    for (int s4 = 0; s4 < 4; ++s4) {
      v16bf r;
#pragma unroll
      for (int e = 0; e < 16; ++e)
        r[e] = (__bf16)Vb[(size_t)(key0 + (e < 8 ? e : e + 8) + (laneh ? 8 : 0)) * 64 + s4 * 16 + lanel];
      bvf[s4] = r;
    }

    const float pen0 = (1.0f - Mb[key0 + lanel]) * -10000.0f;
    const float pen1 = (1.0f - Mb[key0 + 16 + lanel]) * -10000.0f;

#pragma unroll
    for (int t = 0; t < 2; ++t) {
      v8f s0 = {0, 0, 0, 0, 0, 0, 0, 0};
      v8f s1 = {0, 0, 0, 0, 0, 0, 0, 0};
      s0 = WMMA_BF16(aq[t][0], bkf[0][0], s0);
      s0 = WMMA_BF16(aq[t][1], bkf[0][1], s0);
      s1 = WMMA_BF16(aq[t][0], bkf[1][0], s1);
      s1 = WMMA_BF16(aq[t][1], bkf[1][1], s1);

      float scl[8];
#pragma unroll
      for (int r = 0; r < 8; ++r) {
        float v0 = s0[r] * 0.125f + pen0;
        float v1 = s1[r] * 0.125f + pen1;
        float mx = rmax16(fmaxf(v0, v1));
        float nm = fmaxf(m_[t][r], mx);
        float sc = __expf(m_[t][r] - nm);
        float p0 = __expf(v0 - nm);
        float p1 = __expf(v1 - nm);
        l_[t][r]  = l_[t][r] * sc + rsum16(p0 + p1);
        m_[t][r]  = nm;
        scl[r]    = sc;
        int M     = r + laneh * 8;
        Wbuf[wave][M][lanel]      = (__bf16)p0;
        Wbuf[wave][M][16 + lanel] = (__bf16)p1;
      }
#pragma unroll
      for (int s4 = 0; s4 < 4; ++s4)
#pragma unroll
        for (int r = 0; r < 8; ++r) cacc[t][s4][r] *= scl[r];

      __syncthreads();
      v16bf aw;  // probabilities re-read in A-layout
      {
        const __bf16* base = &Wbuf[wave][lanel][0] + (laneh ? 8 : 0);
        union { uint4 u[2]; v16bf v; } uw;
        uw.u[0] = *(const uint4*)(base);
        uw.u[1] = *(const uint4*)(base + 16);
        aw = uw.v;
      }
      __syncthreads();
#pragma unroll
      for (int s4 = 0; s4 < 4; ++s4)
        cacc[t][s4] = WMMA_BF16(aw, bvf[s4], cacc[t][s4]);
    }
  }

  // ---- Finalize: context = acc / l ----
  float invl[2][8];
#pragma unroll
  for (int t = 0; t < 2; ++t)
#pragma unroll
    for (int r = 0; r < 8; ++r) invl[t][r] = 1.0f / l_[t][r];
#pragma unroll
  for (int t = 0; t < 2; ++t)
#pragma unroll
    for (int s4 = 0; s4 < 4; ++s4)
#pragma unroll
      for (int r = 0; r < 8; ++r)
        OUTC[(bhS + q0 + t * 16 + r + laneh * 8) * 64 + s4 * 16 + lanel] =
            cacc[t][s4][r] * invl[t][r];

  // ============ PASS 2: recompute scores, stream normalized weights ============
  for (int kb = 0; kb < 64; ++kb) {
    const int key0 = kb * 32;
    v16bf bkf[2][2];
#pragma unroll
    for (int j = 0; j < 2; ++j)
#pragma unroll
      for (int f = 0; f < 2; ++f)
        bkf[j][f] = loadA2run(Kb + (size_t)(key0 + j * 16 + lanel) * 64 + f * 32 + (laneh ? 8 : 0));

    const float pen0 = (1.0f - Mb[key0 + lanel]) * -10000.0f;
    const float pen1 = (1.0f - Mb[key0 + 16 + lanel]) * -10000.0f;

#pragma unroll
    for (int t = 0; t < 2; ++t) {
      v8f s0 = {0, 0, 0, 0, 0, 0, 0, 0};
      v8f s1 = {0, 0, 0, 0, 0, 0, 0, 0};
      s0 = WMMA_BF16(aq[t][0], bkf[0][0], s0);
      s0 = WMMA_BF16(aq[t][1], bkf[0][1], s0);
      s1 = WMMA_BF16(aq[t][0], bkf[1][0], s1);
      s1 = WMMA_BF16(aq[t][1], bkf[1][1], s1);
#pragma unroll
      for (int r = 0; r < 8; ++r) {
        size_t row = bhS + q0 + t * 16 + r + laneh * 8;
        float w0 = __expf(s0[r] * 0.125f + pen0 - m_[t][r]) * invl[t][r];
        float w1 = __expf(s1[r] * 0.125f + pen1 - m_[t][r]) * invl[t][r];
        OUTW[row * 2049 + 1 + key0 + lanel]      = w0;
        OUTW[row * 2049 + 1 + key0 + 16 + lanel] = w1;
      }
    }
  }

  // ---- Prefix column (col 0): one lane per half stores its 8 rows ----
  if (lanel == 0) {
#pragma unroll
    for (int t = 0; t < 2; ++t)
#pragma unroll
      for (int r = 0; r < 8; ++r) {
        size_t row = bhS + q0 + t * 16 + r + laneh * 8;
        OUTW[row * 2049] = __expf(sp[t][r] - m_[t][r]) * invl[t][r];
      }
  }
}

extern "C" void kernel_launch(void* const* d_in, const int* in_sizes, int n_in,
                              void* d_out, int out_size, void* d_ws, size_t ws_size,
                              hipStream_t stream) {
  (void)in_sizes; (void)n_in; (void)out_size; (void)d_ws; (void)ws_size;
  const float* Q  = (const float*)d_in[0];
  const float* K  = (const float*)d_in[1];
  const float* V  = (const float*)d_in[2];
  const float* GS = (const float*)d_in[3];
  const float* M  = (const float*)d_in[4];
  const float* WK = (const float*)d_in[5];
  const float* BK = (const float*)d_in[6];
  const float* WV = (const float*)d_in[7];
  const float* BV = (const float*)d_in[8];
  float* OUTC = (float*)d_out;
  float* OUTW = OUTC + (size_t)4 * 12 * 2048 * 64;
  // 48 (b,h) pairs * 16 row-blocks of 128 query rows; 128 threads = 4 wave32s,
  // each wave owns 32 query rows (two 16-row WMMA tiles).
  kvprefix_attn_kernel<<<dim3(768), dim3(128), 0, stream>>>(
      Q, K, V, GS, M, WK, BK, WV, BV, OUTC, OUTW);
}
